// generator_32804960207388
// MI455X (gfx1250) — matrix operand order repaired and checked by reading, not compiled, not run on hardware
//
#include <hip/hip_runtime.h>
#include <math.h>

// Problem constants (match reference)
#define NN 8
#define PP 4096
#define BB 8
#define ZZ 128
#define DF 256
#define TP 64      // points per workgroup tile
#define TPPAD 72   // padded LDS row stride in halves (144B = 16B aligned)

typedef __attribute__((ext_vector_type(16))) _Float16 v16h;
typedef __attribute__((ext_vector_type(8)))  _Float16 v8h;
typedef __attribute__((ext_vector_type(8)))  float    v8f;
typedef __attribute__((ext_vector_type(4)))  float    v4f;

__device__ __forceinline__ float applyAct(float y, int act) {
    if (act == 1) return y > 0.f ? y : 0.02f * y;   // leaky_relu(0.02)
    if (act == 2) return y > 0.f ? y : 0.f;         // relu
    return y;
}

// Y[Mout x TP] = act( W[Mout x Kin] @ X[Kin x TP] + bias )
// X is LDS f16 [row * TPPAD + col]; W,bias are global f32.
// If S != nullptr, results are stored as f32 into S[M*TP + col] instead (Mout <= 16).
__device__ __forceinline__ void gemm16(const float* __restrict__ W,
                                       const float* __restrict__ bias,
                                       int Mout, int Kin,
                                       const _Float16* __restrict__ X,
                                       _Float16* __restrict__ Y,
                                       float* __restrict__ S,
                                       int act, int tid)
{
    const int wave = tid >> 5, lane = tid & 31;
    const int m16  = lane & 15, hi = lane >> 4, ko = hi << 3;
    const int Ksteps = (Kin + 31) >> 5;

    if (Mout >= 64) {
        // ---- blocked path: wave owns M-tiles {wave, wave+8}; 4 nt accumulators
        const int MT = Mout >> 4;
        for (int mt = wave; mt < MT; mt += 8) {
            const int mrow = mt * 16 + m16;          // always < Mout here
            v8f c[4];
            #pragma unroll
            for (int r = 0; r < 8; ++r) {
                float bv = bias[mt * 16 + r + (hi << 3)];
                #pragma unroll
                for (int nt = 0; nt < 4; ++nt) c[nt][r] = bv;
            }
            for (int kt = 0; kt < Ksteps; ++kt) {
                const int kb = kt << 5;
                union { v16h v; _Float16 e[16]; } a;
                if ((Kin & 31) == 0) {
                    // fully in-bounds, 16B-aligned: 4 x global_load_b128
                    const float* wr = W + mrow * Kin + kb;
                    v4f w0 = *(const v4f*)(wr + ko);
                    v4f w1 = *(const v4f*)(wr + ko + 4);
                    v4f w2 = *(const v4f*)(wr + ko + 16);
                    v4f w3 = *(const v4f*)(wr + ko + 20);
                    #pragma unroll
                    for (int j = 0; j < 4; ++j) {
                        a.e[j]      = (_Float16)w0[j];
                        a.e[4 + j]  = (_Float16)w1[j];
                        a.e[8 + j]  = (_Float16)w2[j];
                        a.e[12 + j] = (_Float16)w3[j];
                    }
                } else {
                    #pragma unroll
                    for (int g = 0; g < 2; ++g) {
                        #pragma unroll
                        for (int j = 0; j < 8; ++j) {
                            int K = kb + ko + (g << 4) + j;
                            float w = (K < Kin) ? W[mrow * Kin + K] : 0.f;
                            a.e[(g << 3) + j] = (_Float16)w;
                        }
                    }
                }
                // 4 back-to-back WMMAs sharing the A fragment
                #pragma unroll
                for (int nt = 0; nt < 4; ++nt) {
                    union { v16h v; _Float16 e[16]; } bm;
                    const _Float16* bp = X + (kb + ko) * TPPAD + nt * 16 + m16;
                    #pragma unroll
                    for (int s = 0; s < 16; ++s) bm.e[s] = bp[(s + ((s >> 3) << 3)) * TPPAD];
                    c[nt] = __builtin_amdgcn_wmma_f32_16x16x32_f16(
                                false, a.v, false, bm.v, (short)0, c[nt], false, false);
                }
            }
            #pragma unroll
            for (int nt = 0; nt < 4; ++nt) {
                #pragma unroll
                for (int r = 0; r < 8; ++r) {
                    int M = mt * 16 + r + (hi << 3);
                    float y = applyAct(c[nt][r], act);
                    Y[M * TPPAD + nt * 16 + m16] = (_Float16)y;
                }
            }
        }
    } else {
        // ---- flat path for tiny output heads (Mout = 3 or 1): 4 tiles
        for (int nt = wave; nt < 4; nt += 8) {
            v8f c;
            #pragma unroll
            for (int r = 0; r < 8; ++r) {
                int M = r + (hi << 3);
                c[r] = (M < Mout) ? bias[M] : 0.f;
            }
            for (int kt = 0; kt < Ksteps; ++kt) {
                const int kb = kt << 5;
                union { v16h v; _Float16 e[16]; } a;
                #pragma unroll
                for (int g = 0; g < 2; ++g) {
                    #pragma unroll
                    for (int j = 0; j < 8; ++j) {
                        int K = kb + ko + (g << 4) + j;
                        float w = (m16 < Mout && K < Kin) ? W[m16 * Kin + K] : 0.f;
                        a.e[(g << 3) + j] = (_Float16)w;
                    }
                }
                union { v16h v; _Float16 e[16]; } bm;
                const _Float16* bp = X + (kb + ko) * TPPAD + nt * 16 + m16;
                #pragma unroll
                for (int s = 0; s < 16; ++s) bm.e[s] = bp[(s + ((s >> 3) << 3)) * TPPAD];
                c = __builtin_amdgcn_wmma_f32_16x16x32_f16(
                        false, a.v, false, bm.v, (short)0, c, false, false);
            }
            #pragma unroll
            for (int r = 0; r < 8; ++r) {
                int M = r + (hi << 3);
                if (M < Mout) {
                    float y = applyAct(c[r], act);
                    if (S) S[M * TP + nt * 16 + m16] = y;
                    else   Y[M * TPPAD + nt * 16 + m16] = (_Float16)y;
                }
            }
        }
    }
}

__global__ __launch_bounds__(256) void geo_fused_kernel(
    const float* __restrict__ points, const float* __restrict__ zs,
    const float* __restrict__ matrices,
    const float* __restrict__ dW1,  const float* __restrict__ db1,
    const float* __restrict__ dW2,  const float* __restrict__ db2,
    const float* __restrict__ dW3,  const float* __restrict__ db3,
    const float* __restrict__ gWd1, const float* __restrict__ gbd1,
    const float* __restrict__ gWd2, const float* __restrict__ gbd2,
    const float* __restrict__ gWd3, const float* __restrict__ gbd3,
    const float* __restrict__ gW1,  const float* __restrict__ gb1,
    const float* __restrict__ gW2,  const float* __restrict__ gb2,
    const float* __restrict__ gW3,  const float* __restrict__ gb3,
    float* __restrict__ out)
{
    extern __shared__ char smem[];
    _Float16* act0 = (_Float16*)(smem);              // 256 x TPPAD f16
    _Float16* act1 = (_Float16*)(smem + 36864);      // 256 x TPPAD f16
    float* psb  = (float*)(smem + 73728);            // 3 x TP
    float* offb = psb  + 3 * TP;                     // 3 x TP
    float* defb = offb + 3 * TP;                     // 3 x TP
    float* sbuf = defb + 3 * TP;                     // 3 x TP

    const int tid = threadIdx.x;
    const int blk = blockIdx.x;
    const int n  = blk >> 6;           // 64 point-tiles per sample
    const int p0 = (blk & 63) * TP;

    float* out_sum = out;                              // N*P
    float* out_max = out + NN * PP;                    // N*P
    float* out_aff = out + 2 * NN * PP;                // N*B*12
    float* out_do  = out + 2 * NN * PP + NN * BB * 12; // N*B*3*P

    // affine_mat = zs[:, :, Z:Z+12] - 0.5  (once per sample, by tile 0)
    if (p0 == 0 && tid < BB * 12) {
        int b = tid / 12, j = tid % 12;
        out_aff[(n * BB + b) * 12 + j] = zs[(n * BB + b) * (ZZ + 12) + ZZ + j] - 0.5f;
    }

    float sumAcc = 0.f, maxAcc = 0.f;   // outm >= 0 always (sigmoid * U[0,1))

    for (int b = 0; b < BB; ++b) {
        const float* zb = zs + (n * BB + b) * (ZZ + 12);
        const float  matv = matrices[n * BB + b];

        // --- stage 0: ps = A @ [pts;1] + pts -----------------------------
        if (tid < TP) {
            int p = p0 + tid;
            float x = points[(n * PP + p) * 3 + 0];
            float y = points[(n * PP + p) * 3 + 1];
            float z = points[(n * PP + p) * 3 + 2];
            #pragma unroll
            for (int i = 0; i < 3; ++i) {
                float a0 = zb[ZZ + i * 4 + 0] - 0.5f;
                float a1 = zb[ZZ + i * 4 + 1] - 0.5f;
                float a2 = zb[ZZ + i * 4 + 2] - 0.5f;
                float a3 = zb[ZZ + i * 4 + 3] - 0.5f;
                float pi = (i == 0) ? x : ((i == 1) ? y : z);
                psb[i * TP + tid] = a0 * x + a1 * y + a2 * z + a3 + pi;
            }
        }
        __syncthreads();
        // fill act0: rows 0..2 = ps, rows 3..130 = z broadcast, 131..159 = 0
        for (int idx = tid; idx < 160 * TP; idx += 256) {
            int row = idx >> 6, col = idx & (TP - 1);
            float v = (row < 3) ? psb[row * TP + col]
                    : ((row < 3 + ZZ) ? zb[row - 3] : 0.f);
            act0[row * TPPAD + col] = (_Float16)v;
        }
        __syncthreads();

        // --- deform MLP --------------------------------------------------
        gemm16(dW1 + b * DF * (ZZ + 3), db1 + b * DF, DF, ZZ + 3, act0, act1, nullptr, 1, tid);
        __syncthreads();
        gemm16(dW2 + b * DF * DF, db2 + b * DF, DF, DF, act1, act0, nullptr, 1, tid);
        __syncthreads();
        gemm16(dW3 + b * 3 * DF, db3 + b * 3, 3, DF, act0, nullptr, sbuf, 0, tid);
        __syncthreads();
        // offsets, deformed = ps + offsets; act1 rows 0..2 = deformed, 3..31 = 0
        for (int idx = tid; idx < 32 * TP; idx += 256) {
            int row = idx >> 6, col = idx & (TP - 1);
            if (row < 3) {
                float off = sbuf[row * TP + col];
                offb[row * TP + col] = off;
                float de = psb[row * TP + col] + off;
                defb[row * TP + col] = de;
                act1[row * TPPAD + col] = (_Float16)de;
            } else {
                act1[row * TPPAD + col] = (_Float16)0.f;
            }
        }
        __syncthreads();

        // --- generator trunk with residual -------------------------------
        gemm16(gWd1 + b * DF * 3, gbd1 + b * DF, DF, 3, act1, act0, nullptr, 1, tid);
        __syncthreads();
        gemm16(gWd2 + b * DF * DF, gbd2 + b * DF, DF, DF, act0, act1, nullptr, 1, tid);
        __syncthreads();
        gemm16(gWd3 + b * 3 * DF, gbd3 + b * 3, 3, DF, act1, nullptr, sbuf, 0, tid);
        __syncthreads();
        // o = gconv + deformed ; act0 rows 0..2 = o, 3..31 = 0
        for (int idx = tid; idx < 32 * TP; idx += 256) {
            int row = idx >> 6, col = idx & (TP - 1);
            float v = (row < 3) ? (sbuf[row * TP + col] + defb[row * TP + col]) : 0.f;
            act0[row * TPPAD + col] = (_Float16)v;
        }
        __syncthreads();

        // --- occupancy head ----------------------------------------------
        gemm16(gW1 + b * DF * 3, gb1 + b * DF, DF, 3, act0, act1, nullptr, 2, tid);
        __syncthreads();
        gemm16(gW2 + b * DF * DF, gb2 + b * DF, DF, DF, act1, act0, nullptr, 2, tid);
        __syncthreads();
        gemm16(gW3 + b * DF, gb3 + b, 1, DF, act0, nullptr, sbuf, 0, tid);
        __syncthreads();

        // --- epilogue: sigmoid, branch reduction, deformed_offsets --------
        if (tid < TP) {
            float s  = 1.f / (1.f + __expf(-sbuf[tid]));
            float om = s * matv;
            sumAcc += om;
            maxAcc = fmaxf(maxAcc, om);
            #pragma unroll
            for (int i = 0; i < 3; ++i)
                out_do[((size_t)(n * BB + b) * 3 + i) * PP + p0 + tid] =
                    offb[i * TP + tid] * s;
        }
        __syncthreads();
    }

    if (tid < TP) {
        out_sum[n * PP + p0 + tid] = sumAcc;
        out_max[n * PP + p0 + tid] = maxAcc;
    }
}

extern "C" void kernel_launch(void* const* d_in, const int* in_sizes, int n_in,
                              void* d_out, int out_size, void* d_ws, size_t ws_size,
                              hipStream_t stream) {
    (void)in_sizes; (void)n_in; (void)d_ws; (void)ws_size; (void)out_size;
    const float* points   = (const float*)d_in[0];
    const float* zs       = (const float*)d_in[1];
    const float* matrices = (const float*)d_in[2];
    const float* dW1  = (const float*)d_in[3];
    const float* db1  = (const float*)d_in[4];
    const float* dW2  = (const float*)d_in[5];
    const float* db2  = (const float*)d_in[6];
    const float* dW3  = (const float*)d_in[7];
    const float* db3  = (const float*)d_in[8];
    const float* gWd1 = (const float*)d_in[9];
    const float* gbd1 = (const float*)d_in[10];
    const float* gWd2 = (const float*)d_in[11];
    const float* gbd2 = (const float*)d_in[12];
    const float* gWd3 = (const float*)d_in[13];
    const float* gbd3 = (const float*)d_in[14];
    const float* gW1  = (const float*)d_in[15];
    const float* gb1  = (const float*)d_in[16];
    const float* gW2  = (const float*)d_in[17];
    const float* gb2  = (const float*)d_in[18];
    const float* gW3  = (const float*)d_in[19];
    const float* gb3  = (const float*)d_in[20];

    const size_t smem = 2u * 256u * TPPAD * sizeof(_Float16)   // 73728
                      + 4u * 3u * TP * sizeof(float);          // + 3072 = 76800
    dim3 grid(NN * (PP / TP));   // 8 * 64 = 512 workgroups
    dim3 block(256);             // 8 wave32 waves
    geo_fused_kernel<<<grid, block, smem, stream>>>(
        points, zs, matrices,
        dW1, db1, dW2, db2, dW3, db3,
        gWd1, gbd1, gWd2, gbd2, gWd3, gbd3,
        gW1, gb1, gW2, gb2, gW3, gb3,
        (float*)d_out);
}
